// MultiHeadAttention_14396730376956
// MI455X (gfx1250) — compile-verified, matrix operand order checked by reading, not run on hardware
//
#include <hip/hip_runtime.h>

// ---------------------------------------------------------------------------
// MultiHeadAttention for MI455X (gfx1250, wave32, WMMA)
// S=4096, B=2, E=768, H=12, D=64.  All matmuls in bf16 WMMA (f32 accum).
// Data movement: GLOBAL_LOAD_ASYNC_TO_LDS_B128 (ASYNCcnt), double-buffered
// in the GEMM so the DMA for tile k+1 overlaps the WMMAs of tile k.
// ---------------------------------------------------------------------------

typedef __bf16 bf16;
typedef __attribute__((ext_vector_type(16))) __bf16 v16bf;
typedef __attribute__((ext_vector_type(8)))  __bf16 v8bf;
typedef __attribute__((ext_vector_type(8)))  float   v8f;
typedef __attribute__((ext_vector_type(4)))  int     i32x4;

#define S_LEN 4096
#define BATCH 2
#define EMB   768
#define NHEAD 12
#define HDIM  64
#define MROWS (S_LEN * BATCH)   // 8192

// ---- async global->LDS copy (CDNA5), guarded so compile cannot regress ----
#if __has_builtin(__builtin_amdgcn_global_load_async_to_lds_b128)
#define USE_ASYNC_LDS 1
// builtin signature (from hipcc diagnostic): param0 = <4 x i32> addrspace(1)*,
// param1 = <4 x i32> addrspace(3)*, then imm offset, imm cpol.
typedef __attribute__((address_space(1))) i32x4 gi32x4;
typedef __attribute__((address_space(3))) i32x4 li32x4;
__device__ __forceinline__ void async_b128(const void* g, void* l) {
    // AS1 generic pointers are bit-identical; AS3 offset = low 32 bits of the
    // generic LDS address (aperture in the high half).
    __builtin_amdgcn_global_load_async_to_lds_b128(
        (gi32x4*)(uintptr_t)g, (li32x4*)(uint32_t)(uintptr_t)l, 0, 0);
}
#else
#define USE_ASYNC_LDS 0
#endif

// Wait until at most N async loads remain outstanding (in-order retirement:
// asynccnt<=N with N newest in flight  =>  everything older has landed).
template <int N>
__device__ __forceinline__ void wait_async_n() {
#if USE_ASYNC_LDS
#if __has_builtin(__builtin_amdgcn_s_wait_asynccnt)
    __builtin_amdgcn_s_wait_asynccnt(N);
#else
    asm volatile("s_wait_asynccnt %0" ::"i"(N) : "memory");
#endif
#endif
}

__device__ __forceinline__ v16bf mk16(v8bf lo, v8bf hi) {
    return __builtin_shufflevector(lo, hi, 0,1,2,3,4,5,6,7,8,9,10,11,12,13,14,15);
}

__device__ __forceinline__ v8f wmma_bf16(v16bf a, v16bf b, v8f c) {
    // (neg_a, A, neg_b, B, c_mod, C, reuse_a, reuse_b)
    return __builtin_amdgcn_wmma_f32_16x16x32_bf16(false, a, false, b, (short)0, c,
                                                   false, false);
}

// ---------------------------------------------------------------------------
// fp32 [S,B,E] -> bf16 [B,S,E]  (row content identical, batch moved outermost)
// ---------------------------------------------------------------------------
__global__ __launch_bounds__(256) void cvt_transpose(const float* __restrict__ src,
                                                     bf16* __restrict__ dst) {
    const int E4 = EMB / 4;
    size_t i = (size_t)blockIdx.x * blockDim.x + threadIdx.x;
    size_t total = (size_t)S_LEN * BATCH * E4;
    if (i >= total) return;
    int e  = (int)(i % E4) * 4;
    size_t t = i / E4;              // t = s*BATCH + b  (src row)
    int b = (int)(t % BATCH);
    int s = (int)(t / BATCH);
    const float4 v = *(const float4*)&src[t * EMB + e];
    bf16* p = &dst[((size_t)b * S_LEN + s) * EMB + e];
    p[0] = (bf16)v.x; p[1] = (bf16)v.y; p[2] = (bf16)v.z; p[3] = (bf16)v.w;
}

// fp32 -> bf16, same layout (weights)
__global__ __launch_bounds__(256) void cvt_flat(const float* __restrict__ src,
                                                bf16* __restrict__ dst, int n4) {
    int i = blockIdx.x * blockDim.x + threadIdx.x;
    if (i >= n4) return;
    const float4 v = *(const float4*)&src[(size_t)i * 4];
    bf16* p = &dst[(size_t)i * 4];
    p[0] = (bf16)v.x; p[1] = (bf16)v.y; p[2] = (bf16)v.z; p[3] = (bf16)v.w;
}

// ---------------------------------------------------------------------------
// GEMM: out[m,n] = sum_k X[m,k] * W[n,k] + bias[n]
//   X: bf16 [M, 768] row-major   W: bf16 [768, 768] row-major (torch [out,in])
// MODE 0: out = bf16 [B,H,S,D] scatter (QKV projections)
// MODE 1: out = fp32 [S,B,E]           (final projection)
// Block: 128 thr (4 waves). Tile 128x128x64, double-buffered LDS (64 KB).
// Wave tile 64x64 (4x4 accs): 32 WMMA per 32 ds_load_b128 per wave per phase.
// ---------------------------------------------------------------------------
__device__ __forceinline__ void stage_tile(const bf16* __restrict__ X,
                                           const bf16* __restrict__ W,
                                           bf16 (*lA)[64], bf16 (*lB)[64],
                                           int tid, int tileM, int tileN, int k0) {
    #pragma unroll
    for (int i = 0; i < 8; ++i) {
        int c  = tid + i * 128;
        int r  = c >> 3;
        int cc = (c & 7) * 8;
        const bf16* gx = &X[(size_t)(tileM + r) * EMB + k0 + cc];
        const bf16* gw = &W[(size_t)(tileN + r) * EMB + k0 + cc];
#if USE_ASYNC_LDS
        async_b128(gx, &lA[r][cc]);
        async_b128(gw, &lB[r][cc]);
#else
        *(v8bf*)&lA[r][cc] = *(const v8bf*)gx;
        *(v8bf*)&lB[r][cc] = *(const v8bf*)gw;
#endif
    }
}

template <int MODE>
__global__ __launch_bounds__(128) void gemm_bf16(const bf16* __restrict__ X,
                                                 const bf16* __restrict__ W,
                                                 const float* __restrict__ bias,
                                                 void* __restrict__ out) {
    __shared__ __align__(16) bf16 lA[2][128][64];   // 32 KB (ping-pong)
    __shared__ __align__(16) bf16 lB[2][128][64];   // 32 KB (ping-pong)

    const int tid  = threadIdx.x;
    const int lane = tid & 31;
    const int wid  = tid >> 5;
    const int l15  = lane & 15;
    const int hl   = lane >> 4;            // 0 / 1 (lane half)
    const int tileM = blockIdx.x * 128;
    const int tileN = blockIdx.y * 128;
    const int waveM = (wid & 1) * 64;
    const int waveN = (wid >> 1) * 64;

    v8f acc[4][4];
    #pragma unroll
    for (int mt = 0; mt < 4; ++mt)
        #pragma unroll
        for (int nt = 0; nt < 4; ++nt) acc[mt][nt] = (v8f){};

    // prologue: kick off tile 0 DMA
    stage_tile(X, W, lA[0], lB[0], tid, tileM, tileN, 0);

    for (int k0 = 0; k0 < EMB; k0 += 64) {
        const int cur = (k0 >> 6) & 1;
        if (k0 + 64 < EMB) {
            // issue tile k+1 DMA into the other buffer, then wait only for
            // tile k (16 newest ops may stay in flight behind the WMMAs)
            stage_tile(X, W, lA[cur ^ 1], lB[cur ^ 1], tid, tileM, tileN, k0 + 64);
            wait_async_n<16>();
        } else {
            wait_async_n<0>();
        }
        __syncthreads();                    // tile k visible to all waves

        #pragma unroll
        for (int ks = 0; ks < 64; ks += 32) {
            v16bf af[4], bfr[4];
            #pragma unroll
            for (int mt = 0; mt < 4; ++mt) {
                int r = waveM + mt * 16 + l15;
                af[mt] = mk16(*(v8bf*)&lA[cur][r][ks + hl * 8],
                              *(v8bf*)&lA[cur][r][ks + 16 + hl * 8]);
            }
            #pragma unroll
            for (int nt = 0; nt < 4; ++nt) {
                int r = waveN + nt * 16 + l15;
                bfr[nt] = mk16(*(v8bf*)&lB[cur][r][ks + hl * 8],
                               *(v8bf*)&lB[cur][r][ks + 16 + hl * 8]);
            }
            #pragma unroll
            for (int mt = 0; mt < 4; ++mt)
                #pragma unroll
                for (int nt = 0; nt < 4; ++nt)
                    acc[mt][nt] = wmma_bf16(af[mt], bfr[nt], acc[mt][nt]);
        }
        __syncthreads();                    // all waves done reading tile k
    }

    // Epilogue. C layout: VGPR r -> M = r + 8*half, N = lane&15 within tile.
    #pragma unroll
    for (int mt = 0; mt < 4; ++mt) {
        #pragma unroll
        for (int nt = 0; nt < 4; ++nt) {
            const int n = tileN + waveN + nt * 16 + l15;
            const float bn = bias[n];
            #pragma unroll
            for (int r = 0; r < 8; ++r) {
                const int m = tileM + waveM + mt * 16 + r + hl * 8;
                const float v = acc[mt][nt][r] + bn;
                const int b = m >> 12, s = m & (S_LEN - 1);
                if (MODE == 0) {
                    const int h = n >> 6, d = n & 63;
                    ((bf16*)out)[((((size_t)b * NHEAD + h) * S_LEN + s) << 6) + d] =
                        (bf16)v;
                } else {
                    ((float*)out)[((size_t)s * BATCH + b) * EMB + n] = v;
                }
            }
        }
    }
}

// ---------------------------------------------------------------------------
// Flash attention: per (b,h): ctx = softmax(Q K^T / 8) V
//   Qp/Kp/Vp: bf16 [B,H,S,D].  ctx: bf16 [B,S,E] (head-concat).
// Block = 128 thr (4 waves), 64 query rows / block, key blocks of 64.
// ---------------------------------------------------------------------------
__global__ __launch_bounds__(128) void flash_attn(const bf16* __restrict__ Qp,
                                                  const bf16* __restrict__ Kp,
                                                  const bf16* __restrict__ Vp,
                                                  bf16* __restrict__ ctx) {
    __shared__ __align__(16) bf16 lK[64][64];        // key-major   8 KB
    __shared__ __align__(16) bf16 lVt[64][72];       // dim-major   9 KB (padded)
    __shared__ __align__(16) bf16 lP[4][16][72];     // per-wave P  9 KB (padded)

    const int tid  = threadIdx.x;
    const int lane = tid & 31;
    const int wid  = tid >> 5;
    const int l15  = lane & 15;
    const int hl   = lane >> 4;

    const size_t headOff = (size_t)blockIdx.y * S_LEN * HDIM;
    const bf16* Qh = Qp + headOff;
    const bf16* Kh = Kp + headOff;
    const bf16* Vh = Vp + headOff;
    const int qBase = blockIdx.x * 64;
    const int bIdx = blockIdx.y / NHEAD;
    const int hIdx = blockIdx.y % NHEAD;

    // Load Q fragments for this wave's 16 rows, scale by 1/sqrt(D)=0.125 (exact).
    const int qRow = qBase + wid * 16 + l15;   // A layout: lane&15 = M (both halves)
    v16bf qf[2];
    #pragma unroll
    for (int kc = 0; kc < 2; ++kc) {
        v8bf lo = *(const v8bf*)&Qh[(size_t)qRow * HDIM + kc * 32 + hl * 8];
        v8bf hi = *(const v8bf*)&Qh[(size_t)qRow * HDIM + kc * 32 + 16 + hl * 8];
        v16bf f = mk16(lo, hi);
        #pragma unroll
        for (int i = 0; i < 16; ++i) f[i] = (bf16)((float)f[i] * 0.125f);
        qf[kc] = f;
    }

    v8f o[4];
    #pragma unroll
    for (int nt = 0; nt < 4; ++nt) o[nt] = (v8f){};
    float mrow[8], lrow[8];
    #pragma unroll
    for (int r = 0; r < 8; ++r) { mrow[r] = -3.0e38f; lrow[r] = 0.0f; }

    for (int j = 0; j < S_LEN / 64; ++j) {
        const int jBase = j * 64;
        __syncthreads();                       // prev iteration done with lK/lVt
        // stage K (async, key-major) and V^T (VGPR transpose, dim-major)
        #pragma unroll
        for (int i = 0; i < 4; ++i) {
            int c  = tid + i * 128;
            int r  = c >> 3;                   // key within block
            int cc = (c & 7) * 8;              // dim offset
#if USE_ASYNC_LDS
            async_b128(&Kh[(size_t)(jBase + r) * HDIM + cc], &lK[r][cc]);
#else
            *(v8bf*)&lK[r][cc] = *(const v8bf*)&Kh[(size_t)(jBase + r) * HDIM + cc];
#endif
            v8bf vv = *(const v8bf*)&Vh[(size_t)(jBase + r) * HDIM + cc];
            #pragma unroll
            for (int t = 0; t < 8; ++t) lVt[cc + t][r] = vv[t];
        }
        // prefetch next key/value block toward L2 while we compute this one
        if (j + 1 < S_LEN / 64) {
            __builtin_prefetch(&Kh[(size_t)(jBase + 64 + (tid >> 1)) * HDIM], 0, 1);
            __builtin_prefetch(&Vh[(size_t)(jBase + 64 + (tid >> 1)) * HDIM], 0, 1);
        }
        wait_async_n<0>();
        __syncthreads();

        // scores: S = (Q*0.125) K^T   (4 key tiles x 2 K-dim chunks)
        v8f sc[4];
        #pragma unroll
        for (int nt = 0; nt < 4; ++nt) sc[nt] = (v8f){};
        #pragma unroll
        for (int nt = 0; nt < 4; ++nt) {
            const int kr = nt * 16 + l15;      // key row (B^T row)
            #pragma unroll
            for (int kc = 0; kc < 2; ++kc) {
                v16bf bfr = mk16(*(v8bf*)&lK[kr][kc * 32 + hl * 8],
                                 *(v8bf*)&lK[kr][kc * 32 + 16 + hl * 8]);
                sc[nt] = wmma_bf16(qf[kc], bfr, sc[nt]);
            }
        }

        // online softmax (per row r+8*hl; N spread over 16 lanes of each half)
        #pragma unroll
        for (int r = 0; r < 8; ++r) {
            float mx = fmaxf(fmaxf(sc[0][r], sc[1][r]), fmaxf(sc[2][r], sc[3][r]));
            #pragma unroll
            for (int msk = 1; msk < 16; msk <<= 1) mx = fmaxf(mx, __shfl_xor(mx, msk));
            const float mnew = fmaxf(mrow[r], mx);
            const float alpha = __expf(mrow[r] - mnew);
            mrow[r] = mnew;
            float rs = 0.0f;
            #pragma unroll
            for (int nt = 0; nt < 4; ++nt) {
                float p = __expf(sc[nt][r] - mnew);
                sc[nt][r] = p;
                rs += p;
            }
            #pragma unroll
            for (int msk = 1; msk < 16; msk <<= 1) rs += __shfl_xor(rs, msk);
            lrow[r] = lrow[r] * alpha + rs;
            #pragma unroll
            for (int nt = 0; nt < 4; ++nt) o[nt][r] *= alpha;
        }

        // C-layout -> A-fragment layout via per-wave LDS round trip
        #pragma unroll
        for (int nt = 0; nt < 4; ++nt)
            #pragma unroll
            for (int r = 0; r < 8; ++r)
                lP[wid][r + hl * 8][nt * 16 + l15] = (bf16)sc[nt][r];
        __syncthreads();

        // O += P V  (K-dim = 64 keys -> 2 chunks; N = 64 dims -> 4 tiles)
        #pragma unroll
        for (int kc = 0; kc < 2; ++kc) {
            v16bf pf = mk16(*(v8bf*)&lP[wid][l15][kc * 32 + hl * 8],
                            *(v8bf*)&lP[wid][l15][kc * 32 + 16 + hl * 8]);
            #pragma unroll
            for (int nt = 0; nt < 4; ++nt) {
                const int vr = nt * 16 + l15;  // output dim (B^T row)
                v16bf vf = mk16(*(v8bf*)&lVt[vr][kc * 32 + hl * 8],
                                *(v8bf*)&lVt[vr][kc * 32 + 16 + hl * 8]);
                o[nt] = wmma_bf16(pf, vf, o[nt]);
            }
        }
    }

    // normalize + write ctx[b][s][h*64 + d]
    #pragma unroll
    for (int nt = 0; nt < 4; ++nt) {
        const int d = nt * 16 + l15;
        #pragma unroll
        for (int r = 0; r < 8; ++r) {
            const int srow = qBase + wid * 16 + r + hl * 8;
            const float v = o[nt][r] / lrow[r];
            ctx[((size_t)bIdx * S_LEN + srow) * EMB + hIdx * HDIM + d] = (bf16)v;
        }
    }
}

// ---------------------------------------------------------------------------
extern "C" void kernel_launch(void* const* d_in, const int* in_sizes, int n_in,
                              void* d_out, int out_size, void* d_ws, size_t ws_size,
                              hipStream_t stream) {
    const float* q  = (const float*)d_in[0];
    const float* k  = (const float*)d_in[1];
    const float* v  = (const float*)d_in[2];
    const float* Wq = (const float*)d_in[3];
    const float* bq = (const float*)d_in[4];
    const float* Wk = (const float*)d_in[5];
    const float* bk = (const float*)d_in[6];
    const float* Wv = (const float*)d_in[7];
    const float* bv = (const float*)d_in[8];
    const float* Wo = (const float*)d_in[9];
    const float* bo = (const float*)d_in[10];

    // workspace layout (bf16 elements); total ~81 MB
    const size_t ACT = (size_t)MROWS * EMB;        // 6,291,456
    const size_t WEL = (size_t)EMB * EMB;          //   589,824
    bf16* qb  = (bf16*)d_ws;
    bf16* kb  = qb  + ACT;
    bf16* vb  = kb  + ACT;
    bf16* Qp  = vb  + ACT;
    bf16* Kp  = Qp  + ACT;
    bf16* Vp  = Kp  + ACT;
    bf16* Wqb = Vp  + ACT;
    bf16* Wkb = Wqb + WEL;
    bf16* Wvb = Wkb + WEL;
    bf16* Wob = Wvb + WEL;
    bf16* ctx = qb;   // qb/kb/vb are dead after the projections; reuse qb

    // 1) convert inputs (fp32 [S,B,E] -> bf16 [B,S,E]) and weights
    {
        const int total4 = MROWS * EMB / 4;
        dim3 g((total4 + 255) / 256);
        cvt_transpose<<<g, 256, 0, stream>>>(q, qb);
        cvt_transpose<<<g, 256, 0, stream>>>(k, kb);
        cvt_transpose<<<g, 256, 0, stream>>>(v, vb);
        const int w4 = EMB * EMB / 4;
        dim3 gw((w4 + 255) / 256);
        cvt_flat<<<gw, 256, 0, stream>>>(Wq, Wqb, w4);
        cvt_flat<<<gw, 256, 0, stream>>>(Wk, Wkb, w4);
        cvt_flat<<<gw, 256, 0, stream>>>(Wv, Wvb, w4);
        cvt_flat<<<gw, 256, 0, stream>>>(Wo, Wob, w4);
    }

    // 2) QKV projections -> [B,H,S,D] bf16
    {
        dim3 g(MROWS / 128, EMB / 128);   // 64 x 6
        gemm_bf16<0><<<g, 128, 0, stream>>>(qb, Wqb, bq, (void*)Qp);
        gemm_bf16<0><<<g, 128, 0, stream>>>(kb, Wkb, bk, (void*)Kp);
        gemm_bf16<0><<<g, 128, 0, stream>>>(vb, Wvb, bv, (void*)Vp);
    }

    // 3) flash attention -> ctx bf16 [B,S,E]
    {
        dim3 g(S_LEN / 64, BATCH * NHEAD);   // 64 x 24
        flash_attn<<<g, 128, 0, stream>>>(Qp, Kp, Vp, ctx);
    }

    // 4) output projection -> d_out fp32 [S,B,E]
    {
        dim3 g(MROWS / 128, EMB / 128);
        gemm_bf16<1><<<g, 128, 0, stream>>>(ctx, Wob, bo, d_out);
    }
}
